// MambaExpert_19885698581132
// MI455X (gfx1250) — matrix operand order repaired and checked by reading, not compiled, not run on hardware
//
#include <hip/hip_runtime.h>
#include <hip/hip_bf16.h>
#include <math.h>

// ---------------- problem constants ----------------
#define B_   256
#define L_   512
#define D_   256
#define NL_  2
#define DS_  16
#define DC_  4
#define DI_  512   // E*D
#define DTR_ 16

typedef __bf16 bf16;
typedef bf16  v8bf  __attribute__((ext_vector_type(8)));
typedef bf16  v16bf __attribute__((ext_vector_type(16)));
typedef float v8f   __attribute__((ext_vector_type(8)));

static __device__ __forceinline__ float sigmoidf_(float x) { return 1.0f / (1.0f + __expf(-x)); }
static __device__ __forceinline__ float siluf_(float x)    { return x * sigmoidf_(x); }

// ---------------- embedding gather ----------------
__global__ __launch_bounds__(256)
void embed_kernel(const int* __restrict__ x, const float* __restrict__ emb,
                  float* __restrict__ h)
{
    size_t idx = (size_t)blockIdx.x * 256 + threadIdx.x;   // over B*L*D
    int d   = (int)(idx & (D_ - 1));
    size_t pos = idx >> 8;                                  // D_ == 256
    int tok = x[pos];
    h[idx] = emb[(size_t)tok * D_ + d];
}

// ---------------- tiled bf16-WMMA GEMM  C = A*W (+bias)(+resid)(silu on hi cols) ----
// AMODE 0: A[m,k] plain.   AMODE 1: A[m,k] = A[m,k] * A2[m,k]  (y * silu(r), g pre-silu'd)
// EPI   0: +bias (if non-null)
// EPI   1: +bias, then silu() applied to columns nb >= nsplit  (gate path of in-proj)
// EPI   2: +bias + resid[m,nb]  (residual out-proj, in place on h)
template<int AMODE, int EPI>
__global__ __launch_bounds__(256)
void gemm_bf16_wmma(const float* __restrict__ A, const float* __restrict__ A2,
                    const float* __restrict__ W, const float* __restrict__ bias,
                    const float* __restrict__ resid, float* __restrict__ C,
                    int M, int N, int K, int lda, int lda2, int ldw, int ldc, int nsplit)
{
    __shared__ __align__(16) bf16 As[128][48];   // 128 x 32 tile, padded to 48 (96B rows)
    __shared__ __align__(16) bf16 Bs[32][72];    // 32 x 64 tile, padded to 72 (144B rows)

    const int tid  = threadIdx.x;
    const int m0   = blockIdx.y * 128;
    const int n0   = blockIdx.x * 64;
    const int wid  = tid >> 5;          // 8 waves: 4 (M) x 2 (N)
    const int wm   = wid >> 1;
    const int wn   = wid & 1;
    const int lane = tid & 31;
    const int half = lane >> 4;
    const int l16  = lane & 15;

    v8f acc[2][2];
    {
        v8f z = {0.f,0.f,0.f,0.f,0.f,0.f,0.f,0.f};
        acc[0][0] = z; acc[0][1] = z; acc[1][0] = z; acc[1][1] = z;
    }

    const int nTiles = K >> 5;
    float4 aReg[4];
    float4 bReg[2];

    auto loadA = [&](int kt) {
        const int k0 = kt << 5;
#pragma unroll
        for (int j = 0; j < 4; ++j) {
            int idx = tid + 256 * j;          // 1024 float4-chunks? no: element-quads
            int row = idx >> 3;               // 128 rows
            int c4  = idx & 7;                // 8 quads of 4 floats = 32 K
            const float* p = A + (size_t)(m0 + row) * lda + (k0 + c4 * 4);
            float4 v = *(const float4*)p;
            if (AMODE == 1) {
                const float* p2 = A2 + (size_t)(m0 + row) * lda2 + (k0 + c4 * 4);
                float4 g = *(const float4*)p2;
                v.x *= g.x; v.y *= g.y; v.z *= g.z; v.w *= g.w;
            }
            aReg[j] = v;
        }
    };
    auto loadB = [&](int kt) {
        const int k0 = kt << 5;
#pragma unroll
        for (int j = 0; j < 2; ++j) {
            int idx = tid + 256 * j;
            int row = idx >> 4;               // 32 K-rows
            int c4  = idx & 15;               // 16 quads = 64 N
            int n   = n0 + c4 * 4;
            float4 v = {0.f, 0.f, 0.f, 0.f};
            if (n + 3 < N) v = *(const float4*)(W + (size_t)(k0 + row) * ldw + n);
            bReg[j] = v;
        }
    };
    auto storeLDS = [&]() {
#pragma unroll
        for (int j = 0; j < 4; ++j) {
            int idx = tid + 256 * j;
            int row = idx >> 3;
            int c4  = idx & 7;
            bf16* p = &As[row][c4 * 4];
            p[0] = (bf16)aReg[j].x; p[1] = (bf16)aReg[j].y;
            p[2] = (bf16)aReg[j].z; p[3] = (bf16)aReg[j].w;
        }
#pragma unroll
        for (int j = 0; j < 2; ++j) {
            int idx = tid + 256 * j;
            int row = idx >> 4;
            int c4  = idx & 15;
            bf16* p = &Bs[row][c4 * 4];
            p[0] = (bf16)bReg[j].x; p[1] = (bf16)bReg[j].y;
            p[2] = (bf16)bReg[j].z; p[3] = (bf16)bReg[j].w;
        }
    };

    loadA(0); loadB(0);
    for (int kt = 0; kt < nTiles; ++kt) {
        __syncthreads();                 // previous tile fully consumed
        storeLDS();
        __syncthreads();
        if (kt + 1 < nTiles) {
            loadA(kt + 1); loadB(kt + 1);   // global loads in flight during math
            if (kt + 2 < nTiles)
                __builtin_prefetch(A + (size_t)(m0 + (tid >> 3)) * lda + ((kt + 2) << 5), 0, 1);
        }

        // ----- build fragments from LDS (ISA 7.12.2 layouts) -----
        v16bf af[2], bfr[2];
#pragma unroll
        for (int r = 0; r < 2; ++r) {
            const v8bf lo = *(const v8bf*)&As[wm * 32 + r * 16 + l16][8 * half];
            const v8bf hi = *(const v8bf*)&As[wm * 32 + r * 16 + l16][16 + 8 * half];
            af[r] = __builtin_shufflevector(lo, hi, 0,1,2,3,4,5,6,7,8,9,10,11,12,13,14,15);
        }
#pragma unroll
        for (int c = 0; c < 2; ++c) {
#pragma unroll
            for (int s = 0; s < 16; ++s) {
                const int kk = (s < 8) ? (8 * half + s) : (16 + 8 * half + (s - 8));
                bfr[c][s] = Bs[kk][wn * 32 + c * 16 + l16];
            }
        }
#pragma unroll
        for (int r = 0; r < 2; ++r)
#pragma unroll
            for (int c = 0; c < 2; ++c)
                acc[r][c] = __builtin_amdgcn_wmma_f32_16x16x32_bf16(
                    false, af[r], false, bfr[c], (short)0, acc[r][c], false, false);
    }

    // ----- epilogue -----
#pragma unroll
    for (int r = 0; r < 2; ++r) {
#pragma unroll
        for (int c = 0; c < 2; ++c) {
            const int nb = n0 + wn * 32 + c * 16 + l16;
            if (nb >= N) continue;
#pragma unroll
            for (int i = 0; i < 8; ++i) {
                const int m = m0 + wm * 32 + r * 16 + half * 8 + i;
                float v = acc[r][c][i];
                if (bias) v += bias[nb];
                if (EPI == 2) v += resid[(size_t)m * ldc + nb];
                if (EPI == 1 && nb >= nsplit) v = siluf_(v);
                C[(size_t)m * ldc + nb] = v;
            }
        }
    }
}

// ---------------- depthwise causal conv (DC=4) + SiLU ----------------
// input: xr[:, 0:DI] with row stride 2*DI ; output: xs (row stride DI)
__global__ __launch_bounds__(256)
void conv_silu_kernel(const float* __restrict__ xr, const float* __restrict__ cw,
                      const float* __restrict__ cb, float* __restrict__ xs)
{
    size_t idx = (size_t)blockIdx.x * 256 + threadIdx.x;    // over M*DI
    int d  = (int)(idx & (DI_ - 1));
    size_t bt = idx >> 9;                                   // DI_ == 512
    int t  = (int)(bt & (L_ - 1));

    float w0 = cw[d * DC_ + 0], w1 = cw[d * DC_ + 1], w2 = cw[d * DC_ + 2], w3 = cw[d * DC_ + 3];
    float acc = cb[d];
    const size_t stride = 2 * DI_;
    if (t >= 3) acc += w0 * xr[(bt - 3) * stride + d];
    if (t >= 2) acc += w1 * xr[(bt - 2) * stride + d];
    if (t >= 1) acc += w2 * xr[(bt - 1) * stride + d];
    acc += w3 * xr[bt * stride + d];
    xs[bt * (size_t)DI_ + d] = siluf_(acc);
}

// ---------------- selective scan (fused delta-proj + softplus), in place on xs ----
__global__ __launch_bounds__(512)
void scan_kernel(const float* __restrict__ dtbc, float* __restrict__ xs,
                 const float* __restrict__ dtw_g, const float* __restrict__ dtb_g,
                 const float* __restrict__ Alog_g, const float* __restrict__ Dp_g)
{
    const int b = blockIdx.x;
    const int d = threadIdx.x;          // 0..511 channel

    float dtw[DTR_], Av[DS_], st[DS_];
#pragma unroll
    for (int k = 0; k < DTR_; ++k) dtw[k] = dtw_g[k * DI_ + d];
#pragma unroll
    for (int s = 0; s < DS_; ++s) { Av[s] = -__expf(Alog_g[d * DS_ + s]); st[s] = 0.f; }
    const float dtb = dtb_g[d];
    const float Dpv = Dp_g[d];

    __shared__ float sbc[32][48];       // 32 timesteps of (dt[16] | B[16] | C[16])

    for (int t0 = 0; t0 < L_; t0 += 32) {
        __syncthreads();
#pragma unroll
        for (int j = 0; j < 3; ++j) {   // 32*48 = 1536 = 512*3
            int idx = d + 512 * j;
            int row = idx / 48, col = idx % 48;
            sbc[row][col] = dtbc[((size_t)b * L_ + t0 + row) * 48 + col];
        }
        __syncthreads();

        for (int tc = 0; tc < 32; ++tc) {
            const size_t off = ((size_t)b * L_ + t0 + tc) * DI_ + d;
            const float u = xs[off];
            float dv = dtb;
#pragma unroll
            for (int k = 0; k < DTR_; ++k) dv += dtw[k] * sbc[tc][k];
            const float delta = (dv > 20.f) ? dv : log1pf(__expf(dv));
            const float du = delta * u;
            float y = 0.f;
#pragma unroll
            for (int s = 0; s < DS_; ++s) {
                st[s] = __expf(delta * Av[s]) * st[s] + du * sbc[tc][16 + s];
                y += st[s] * sbc[tc][32 + s];
            }
            xs[off] = y + u * Dpv;      // read-before-write, in place
        }
    }
}

// ---------------- LayerNorm + time-mean + MLP head ----------------
static __device__ __forceinline__ float blockReduceSum256(float v)
{
    __shared__ float wsum[8];
#pragma unroll
    for (int o = 16; o > 0; o >>= 1) v += __shfl_xor(v, o);
    __syncthreads();                    // protect wsum from previous use
    if ((threadIdx.x & 31) == 0) wsum[threadIdx.x >> 5] = v;
    __syncthreads();
    float s = 0.f;
#pragma unroll
    for (int i = 0; i < 8; ++i) s += wsum[i];
    return s;
}

__global__ __launch_bounds__(256)
void head_kernel(const float* __restrict__ h, const float* __restrict__ g,
                 const float* __restrict__ bln, const float* __restrict__ w1,
                 const float* __restrict__ b1, const float* __restrict__ w2,
                 const float* __restrict__ b2, float* __restrict__ out)
{
    const int b = blockIdx.x, tid = threadIdx.x;
    const float gd = g[tid], bd = bln[tid];
    float acc = 0.f;
    for (int t = 0; t < L_; ++t) {
        float v = h[((size_t)b * L_ + t) * D_ + tid];
        float mu  = blockReduceSum256(v) * (1.f / D_);
        float dv  = v - mu;
        float var = blockReduceSum256(dv * dv) * (1.f / D_);
        acc += dv * rsqrtf(var + 1e-5f) * gd + bd;
    }
    __shared__ float p[D_];
    __shared__ float q[D_ / 2];
    p[tid] = acc * (1.f / L_);
    __syncthreads();
    if (tid < D_ / 2) {
        float s = b1[tid];
        for (int dd = 0; dd < D_; ++dd) s += p[dd] * w1[dd * (D_ / 2) + tid];
        q[tid] = fmaxf(s, 0.f);
    }
    __syncthreads();
    if (tid < 2) {
        float s = b2[tid];
        for (int j = 0; j < D_ / 2; ++j) s += q[j] * w2[j * 2 + tid];
        out[b * 2 + tid] = s;
    }
}

// ---------------- launch ----------------
extern "C" void kernel_launch(void* const* d_in, const int* in_sizes, int n_in,
                              void* d_out, int out_size, void* d_ws, size_t ws_size,
                              hipStream_t stream)
{
    (void)in_sizes; (void)n_in; (void)out_size; (void)ws_size;
    const int*   x      = (const int*)  d_in[0];
    const float* emb    = (const float*)d_in[1];
    const float* in_w   = (const float*)d_in[2];
    const float* in_b   = (const float*)d_in[3];
    const float* conv_w = (const float*)d_in[4];
    const float* conv_b = (const float*)d_in[5];
    const float* xproj  = (const float*)d_in[6];
    const float* dt_w   = (const float*)d_in[7];
    const float* dt_b   = (const float*)d_in[8];
    const float* A_log  = (const float*)d_in[9];
    const float* Dp     = (const float*)d_in[10];
    const float* out_w  = (const float*)d_in[11];
    const float* out_b  = (const float*)d_in[12];
    const float* ln_g   = (const float*)d_in[13];
    const float* ln_b   = (const float*)d_in[14];
    const float* w1     = (const float*)d_in[15];
    const float* b1     = (const float*)d_in[16];
    const float* w2     = (const float*)d_in[17];
    const float* b2     = (const float*)d_in[18];
    float* out = (float*)d_out;

    const size_t M = (size_t)B_ * L_;          // 131072
    float* ws   = (float*)d_ws;
    float* hbuf = ws;                          // M*D            (134 MB)
    float* xr   = hbuf + M * D_;               // M*2*DI (xs_raw | silu(r))  (537 MB)
    float* xs   = xr   + M * (2 * DI_);        // M*DI  (conv out -> scan in place) (268 MB)
    float* dtbc = xs   + M * DI_;              // M*48  (dt|B|C) (25 MB)

    embed_kernel<<<dim3((unsigned)(M * D_ / 256)), 256, 0, stream>>>(x, emb, hbuf);

    for (int l = 0; l < NL_; ++l) {
        const float* in_w_l   = in_w   + (size_t)l * D_  * 2 * DI_;
        const float* in_b_l   = in_b   + (size_t)l * 2 * DI_;
        const float* conv_w_l = conv_w + (size_t)l * DI_ * DC_;
        const float* conv_b_l = conv_b + (size_t)l * DI_;
        const float* xproj_l  = xproj  + (size_t)l * DI_ * 48;
        const float* dt_w_l   = dt_w   + (size_t)l * DTR_ * DI_;
        const float* dt_b_l   = dt_b   + (size_t)l * DI_;
        const float* Alog_l   = A_log  + (size_t)l * DI_ * DS_;
        const float* Dp_l     = Dp     + (size_t)l * DI_;
        const float* out_w_l  = out_w  + (size_t)l * DI_ * D_;
        const float* out_b_l  = out_b  + (size_t)l * D_;

        // in-proj: xr = h @ in_w + in_b ; silu applied to gate half (cols >= DI)
        gemm_bf16_wmma<0, 1><<<dim3(2 * DI_ / 64, (unsigned)(M / 128)), 256, 0, stream>>>(
            hbuf, nullptr, in_w_l, in_b_l, nullptr, xr,
            (int)M, 2 * DI_, D_, D_, 0, 2 * DI_, 2 * DI_, DI_);

        // depthwise causal conv + silu: xs = silu(conv(xr[:, :DI]))
        conv_silu_kernel<<<dim3((unsigned)(M * DI_ / 256)), 256, 0, stream>>>(
            xr, conv_w_l, conv_b_l, xs);

        // x-proj: dtbc = xs @ xproj_w   (N = 48 = dt|B|C)
        gemm_bf16_wmma<0, 0><<<dim3(1, (unsigned)(M / 128)), 256, 0, stream>>>(
            xs, nullptr, xproj_l, nullptr, nullptr, dtbc,
            (int)M, 48, DI_, DI_, 0, 48, 48, 0);

        // selective scan (delta-proj + softplus fused), in place: xs <- y
        scan_kernel<<<dim3(B_), DI_, 0, stream>>>(dtbc, xs, dt_w_l, dt_b_l, Alog_l, Dp_l);

        // out-proj with fused gate + residual, in place on h:
        // h = (y * silu(r)) @ out_w + out_b + h
        gemm_bf16_wmma<1, 2><<<dim3(D_ / 64, (unsigned)(M / 128)), 256, 0, stream>>>(
            xs, xr + DI_, out_w_l, out_b_l, hbuf, hbuf,
            (int)M, D_, DI_, DI_, 2 * DI_, D_, D_, 0);
    }

    head_kernel<<<dim3(B_), D_, 0, stream>>>(hbuf, ln_g, ln_b, w1, b1, w2, b2, out);
}
